// DynamicStructuralMasking_22917945492159
// MI455X (gfx1250) — matrix operand order repaired and checked by reading, not compiled, not run on hardware
//
#include <hip/hip_runtime.h>
#include <hip/hip_bf16.h>

#define D_MODEL   1024
#define NUM_HEADS 16
#define D_K       64
#define SEQ       2048
#define BATCH     2
#define ATTN_SCALE 0.125f      // 1/sqrt(64)
#define SIM_THRESH 0.7f

typedef __attribute__((ext_vector_type(16))) __bf16 bf16x16;
typedef __attribute__((ext_vector_type(8)))  __bf16 bf16x8;
typedef __attribute__((ext_vector_type(8)))  float  f32x8;
typedef __attribute__((ext_vector_type(4)))  float  f32x4;
typedef __attribute__((ext_vector_type(4)))  unsigned int u32x4;
typedef __attribute__((ext_vector_type(8)))  unsigned int u32x8;

#define NEG_INF (-__builtin_inff())

// ---- LDS offset of a shared-memory pointer (generic -> AS3 -> int) ------
typedef __attribute__((address_space(3))) const void* lds_cptr_t;
__device__ __forceinline__ unsigned lds_offset_of(const void* p) {
  return (unsigned)(unsigned long long)(lds_cptr_t)p;
}

// ---- Tensor Data Mover: 2D bf16 tile, global(row-major) -> LDS ----------
// D# per CDNA5 ISA 8.3/8.4: group0 = {count, lds_addr, global_addr, type=2},
// group1 = {data_size=2B, tensor_dim0/1, tile_dim0/1, tensor_dim0_stride}.
__device__ __forceinline__ void tdm_load_2d_bf16(unsigned lds_off, const void* gptr,
                                                 unsigned tile0, unsigned tile1,
                                                 unsigned long long stride0_elems) {
  unsigned long long ga = (unsigned long long)gptr;
  u32x4 g0;
  g0[0] = 1u;                                                // count=1, user mode
  g0[1] = lds_off;                                           // lds_addr [63:32]
  g0[2] = (unsigned)ga;                                      // global_addr [95:64]
  g0[3] = (unsigned)((ga >> 32) & 0x01ffffffu) | (2u << 30); // ga[56:32] | type=2
  u32x8 g1;
  g1[0] = 1u << 16;                                          // data_size=1 (2 bytes)
  g1[1] = (tile0 & 0xffffu) << 16;                           // tensor_dim0 lo16
  g1[2] = (tile0 >> 16) | ((tile1 & 0xffffu) << 16);         // tdim0 hi | tensor_dim1 lo
  g1[3] = (tile1 >> 16) | ((tile0 & 0xffffu) << 16);         // tdim1 hi | tile_dim0
  g1[4] = tile1 & 0xffffu;                                   // tile_dim1 ; tile_dim2=0
  g1[5] = (unsigned)stride0_elems;                           // tensor_dim0_stride lo32
  g1[6] = (unsigned)((stride0_elems >> 32) & 0xffffu);       // stride hi16 ; dim1_stride=0
  g1[7] = 0u;
  asm volatile("tensor_load_to_lds %0, %1" :: "s"(g0), "s"(g1) : "memory");
}

// ---- WMMA fragment helpers (CDNA5 16x16x32 bf16, wave32) ----------------
// A 16x32 (MxK): lane(l,half): row=l, K=[16*half..+15]
__device__ __forceinline__ bf16x16 load_a_frag(const __bf16* A, int lda, int l, int half) {
  const __bf16* p = A + (size_t)l * lda + half * 16;
  bf16x8 lo = *(const bf16x8*)(p);
  bf16x8 hi = *(const bf16x8*)(p + 8);
  bf16x16 r;
#pragma unroll
  for (int i = 0; i < 8; ++i) { r[i] = lo[i]; r[i + 8] = hi[i]; }
  return r;
}

// B 32x16 (KxN) given B^T (NxK row-major): lane(l,half): col=l, K=[16*half..+15]
__device__ __forceinline__ bf16x16 load_bt_frag(const __bf16* Bt, int ldb, int l, int half) {
  return *(const bf16x16*)(Bt + (size_t)l * ldb + half * 16);
}

// A fragment sourced from fp32 LDS rows (converted to bf16 on the fly)
__device__ __forceinline__ bf16x16 load_a_frag_lds(const float* P, int ldp, int l, int half, int k0) {
  const float* p = P + (size_t)l * ldp + k0 + half * 16;
  bf16x16 r;
#pragma unroll
  for (int i = 0; i < 8; ++i) { r[i] = (__bf16)p[i]; r[i + 8] = (__bf16)p[8 + i]; }
  return r;
}

__device__ __forceinline__ f32x8 wmma_bf16(bf16x16 a, bf16x16 b, f32x8 c) {
  return __builtin_amdgcn_wmma_f32_16x16x32_bf16(false, a, false, b, (short)0, c, false, false);
}

// ---- K1: row norms + bf16 conversion ------------------------------------
__global__ void __launch_bounds__(256) k_norm(const float* __restrict__ x,
                                              __bf16* __restrict__ xbf,
                                              __bf16* __restrict__ xnbf) {
  int row = blockIdx.x;                    // 0 .. B*S-1
  int tid = threadIdx.x;                   // 256 threads, 4 elems each
  const float* xr = x + (size_t)row * D_MODEL;
  f32x4 v = *(const f32x4*)(xr + tid * 4);
  float ss = v[0]*v[0] + v[1]*v[1] + v[2]*v[2] + v[3]*v[3];
#pragma unroll
  for (int off = 16; off >= 1; off >>= 1) ss += __shfl_xor(ss, off, 32);
  __shared__ float red[8];
  __shared__ float s_inv;
  int wid = tid >> 5, lane = tid & 31;
  if (lane == 0) red[wid] = ss;
  __syncthreads();
  if (tid == 0) {
    float t = 0.f;
#pragma unroll
    for (int i = 0; i < 8; ++i) t += red[i];
    float nrm = fmaxf(sqrtf(t), 1e-12f);
    s_inv = 1.0f / nrm;
  }
  __syncthreads();
  float inv = s_inv;
  __bf16* xo  = xbf  + (size_t)row * D_MODEL + tid * 4;
  __bf16* xno = xnbf + (size_t)row * D_MODEL + tid * 4;
#pragma unroll
  for (int i = 0; i < 4; ++i) { xo[i] = (__bf16)v[i]; xno[i] = (__bf16)(v[i] * inv); }
}

// ---- K1b: fp32 -> bf16 bulk convert (weights) ---------------------------
__global__ void __launch_bounds__(256) k_f2bf(const float* __restrict__ s,
                                              __bf16* __restrict__ d, int n) {
  int i = (blockIdx.x * blockDim.x + threadIdx.x) * 4;
  if (i + 3 < n) {
    f32x4 v = *(const f32x4*)(s + i);
#pragma unroll
    for (int j = 0; j < 4; ++j) d[i + j] = (__bf16)v[j];
  }
}

// ---- K2: Y = X W^T + b. Block: 64 s-rows (TDM-staged) x 256 o-cols ------
// Each wave: 64x64 register tile (4x4 WMMA accumulators), 4 WMMA per B-frag.
// out layout: transV==0 -> [B,H,S,dk] ; transV==1 -> [B,H,dk,S]
__global__ void __launch_bounds__(128) k_proj(const __bf16* __restrict__ xb,
                                              const __bf16* __restrict__ Wb,
                                              const float*  __restrict__ bias,
                                              __bf16* __restrict__ out, int transV) {
  extern __shared__ __bf16 Atile[];              // [64][D_MODEL]  128KB
  int s0 = blockIdx.x * 64;
  int b  = blockIdx.z;
  int tid = threadIdx.x;
  int w = tid >> 5, lane = tid & 31, l = lane & 15, half = lane >> 4;
  int o0 = blockIdx.y * 256 + w * 64;

  const __bf16* Arow = xb + ((size_t)b * SEQ + s0) * D_MODEL;
  if (w == 0) {
    tdm_load_2d_bf16(lds_offset_of(Atile), Arow, D_MODEL, 64, D_MODEL);
    __builtin_amdgcn_s_wait_tensorcnt(0);
  }
  __syncthreads();

  f32x8 acc[4][4];                               // [s-subtile][o-subtile]
#pragma unroll
  for (int i = 0; i < 4; ++i) {
    float bv = bias[o0 + 16 * i + l];
#pragma unroll
    for (int st = 0; st < 4; ++st)
#pragma unroll
      for (int r = 0; r < 8; ++r) acc[st][i][r] = bv;
  }
  for (int k0 = 0; k0 < D_MODEL; k0 += 32) {
    bf16x16 a[4];
#pragma unroll
    for (int st = 0; st < 4; ++st)
      a[st] = load_a_frag(Atile + (size_t)(st * 16) * D_MODEL + k0, D_MODEL, l, half);
#pragma unroll
    for (int i = 0; i < 4; ++i) {
      const __bf16* wrow = Wb + (size_t)(o0 + 16 * i) * D_MODEL + k0;
      __builtin_prefetch(wrow + 64, 0, 1);
      bf16x16 bm = load_bt_frag(wrow, D_MODEL, l, half);
#pragma unroll
      for (int st = 0; st < 4; ++st)
        acc[st][i] = wmma_bf16(a[st], bm, acc[st][i]);
    }
  }
#pragma unroll
  for (int st = 0; st < 4; ++st)
#pragma unroll
    for (int i = 0; i < 4; ++i)
#pragma unroll
      for (int r = 0; r < 8; ++r) {
        int m = st * 16 + r + half * 8;
        int o = o0 + 16 * i + l;
        int h = o >> 6, dk = o & 63;
        size_t idx = transV
          ? (((size_t)b * NUM_HEADS + h) * D_K + dk) * SEQ + (s0 + m)
          : (((size_t)b * NUM_HEADS + h) * SEQ + (s0 + m)) * D_K + dk;
        out[idx] = (__bf16)acc[st][i][r];
      }
}

// ---- K3: sim = xn xn^T, byte mask = sim > 0.7. 64 s-rows staged by TDM --
__global__ void __launch_bounds__(128) k_mask(const __bf16* __restrict__ xnb,
                                              unsigned char* __restrict__ mask) {
  extern __shared__ __bf16 Antile[];             // [64][D_MODEL]  128KB
  int s0 = blockIdx.x * 64;
  int b  = blockIdx.z;
  int tid = threadIdx.x;
  int w = tid >> 5, lane = tid & 31, l = lane & 15, half = lane >> 4;
  int t0 = blockIdx.y * 256 + w * 64;

  const __bf16* base = xnb + (size_t)b * SEQ * D_MODEL;
  if (w == 0) {
    tdm_load_2d_bf16(lds_offset_of(Antile), base + (size_t)s0 * D_MODEL, D_MODEL, 64, D_MODEL);
    __builtin_amdgcn_s_wait_tensorcnt(0);
  }
  __syncthreads();

  f32x8 acc[4][4];
#pragma unroll
  for (int st = 0; st < 4; ++st)
#pragma unroll
    for (int i = 0; i < 4; ++i)
#pragma unroll
      for (int r = 0; r < 8; ++r) acc[st][i][r] = 0.f;

  for (int k0 = 0; k0 < D_MODEL; k0 += 32) {
    bf16x16 a[4];
#pragma unroll
    for (int st = 0; st < 4; ++st)
      a[st] = load_a_frag(Antile + (size_t)(st * 16) * D_MODEL + k0, D_MODEL, l, half);
#pragma unroll
    for (int i = 0; i < 4; ++i) {
      bf16x16 bm = load_bt_frag(base + (size_t)(t0 + 16 * i) * D_MODEL + k0, D_MODEL, l, half);
#pragma unroll
      for (int st = 0; st < 4; ++st)
        acc[st][i] = wmma_bf16(a[st], bm, acc[st][i]);
    }
  }
#pragma unroll
  for (int st = 0; st < 4; ++st)
#pragma unroll
    for (int i = 0; i < 4; ++i)
#pragma unroll
      for (int r = 0; r < 8; ++r) {
        int m = s0 + st * 16 + r + half * 8;
        mask[((size_t)b * SEQ + m) * SEQ + (t0 + 16 * i + l)] =
            (acc[st][i][r] > SIM_THRESH) ? (unsigned char)1 : (unsigned char)0;
      }
}

// ---- K4: attention for one (b,h,16-row tile); scores resident in LDS ----
__global__ void __launch_bounds__(256) k_attn(const __bf16* __restrict__ qb,
                                              const __bf16* __restrict__ kb,
                                              const __bf16* __restrict__ vtb,
                                              const unsigned char* __restrict__ mask,
                                              float* __restrict__ out) {
  extern __shared__ float smem[];
  float* P    = smem;                            // [16][SEQ] scores->probs->partials
  float* rinv = smem + 16 * SEQ;                 // [16]
  __bf16* Qt  = (__bf16*)(smem + 16 * SEQ + 16); // [16][64] TDM-staged Q tile

  int s0 = blockIdx.x * 16;
  int h  = blockIdx.y;
  int b  = blockIdx.z;
  int tid = threadIdx.x;
  int w = tid >> 5, lane = tid & 31, l = lane & 15, half = lane >> 4;

  size_t bh = (size_t)b * NUM_HEADS + h;
  const __bf16* Q  = qb  + (bh * SEQ + s0) * D_K;
  const __bf16* K  = kb  + bh * SEQ * D_K;
  const __bf16* Vt = vtb + bh * D_K * SEQ;

  if (w == 0) {
    tdm_load_2d_bf16(lds_offset_of(Qt), Q, D_K, 16, D_K);
    __builtin_amdgcn_s_wait_tensorcnt(0);
  }
  __syncthreads();

  // phase 1: scores = Q K^T into LDS, wave w owns t in [w*256, w*256+256)
  bf16x16 aq0 = load_a_frag(Qt, D_K, l, half);
  bf16x16 aq1 = load_a_frag(Qt + 32, D_K, l, half);
  int tbase = w * 256;
  for (int tt = 0; tt < 16; ++tt) {
    int t0 = tbase + tt * 16;
    f32x8 c;
#pragma unroll
    for (int r = 0; r < 8; ++r) c[r] = 0.f;
    c = wmma_bf16(aq0, load_bt_frag(K + (size_t)t0 * D_K, D_K, l, half), c);
    c = wmma_bf16(aq1, load_bt_frag(K + (size_t)t0 * D_K + 32, D_K, l, half), c);
#pragma unroll
    for (int r = 0; r < 8; ++r) {
      int m = r + half * 8;
      P[(size_t)m * SEQ + t0 + l] = c[r];
    }
  }
  __syncthreads();

  // phase 2: masked scaled softmax per row; un-normalized probs + 1/sum
  {
    int row = tid >> 4, j = tid & 15;            // 16 threads per row
    float* Pr = P + (size_t)row * SEQ;
    const unsigned char* mr = mask + ((size_t)b * SEQ + s0 + row) * SEQ;
    int c0 = j * 128;
    float mx = NEG_INF;
    for (int t = c0; t < c0 + 128; ++t) {
      float v = mr[t] ? Pr[t] * ATTN_SCALE : NEG_INF;
      Pr[t] = v;
      mx = fmaxf(mx, v);
    }
#pragma unroll
    for (int off = 8; off >= 1; off >>= 1) mx = fmaxf(mx, __shfl_xor(mx, off, 16));
    float sum = 0.f;
    for (int t = c0; t < c0 + 128; ++t) {
      float p = __expf(Pr[t] - mx);
      Pr[t] = p;
      sum += p;
    }
#pragma unroll
    for (int off = 8; off >= 1; off >>= 1) sum += __shfl_xor(sum, off, 16);
    if (j == 0) rinv[row] = 1.0f / sum;          // diag always unmasked -> sum > 0
  }
  __syncthreads();

  // phase 3: O_partial = P[:, tbase:tbase+256] @ V, 4 n-tiles of 16
  f32x8 o[4];
#pragma unroll
  for (int i = 0; i < 4; ++i)
#pragma unroll
    for (int r = 0; r < 8; ++r) o[i][r] = 0.f;
  for (int kk = 0; kk < 8; ++kk) {
    int tk = tbase + kk * 32;
    bf16x16 a = load_a_frag_lds(P, SEQ, l, half, tk);
#pragma unroll
    for (int i = 0; i < 4; ++i) {
      bf16x16 bm = load_bt_frag(Vt + (size_t)(16 * i) * SEQ + tk, SEQ, l, half);
      o[i] = wmma_bf16(a, bm, o[i]);
    }
  }
  __syncthreads();                               // all P reads done; reuse as scratch
  float* part = P;                               // [8][16][64]
#pragma unroll
  for (int i = 0; i < 4; ++i)
#pragma unroll
    for (int r = 0; r < 8; ++r) {
      int m = r + half * 8;
      part[(size_t)w * 1024 + m * 64 + 16 * i + l] = o[i][r];
    }
  __syncthreads();

  float* O = out + (bh * SEQ + s0) * D_K;
  for (int e = tid; e < 1024; e += 256) {
    int m = e >> 6, n = e & 63;
    float s = 0.f;
#pragma unroll
    for (int ww = 0; ww < 8; ++ww) s += part[(size_t)ww * 1024 + e];
    O[(size_t)m * D_K + n] = s * rinv[m];
  }
}

// ---- host launcher -------------------------------------------------------
extern "C" void kernel_launch(void* const* d_in, const int* in_sizes, int n_in,
                              void* d_out, int out_size, void* d_ws, size_t ws_size,
                              hipStream_t stream) {
  (void)in_sizes; (void)n_in; (void)out_size; (void)ws_size;
  const float* x  = (const float*)d_in[0];
  const float* Wq = (const float*)d_in[1];
  const float* bq = (const float*)d_in[2];
  const float* Wk = (const float*)d_in[3];
  const float* bk = (const float*)d_in[4];
  const float* Wv = (const float*)d_in[5];
  const float* bv = (const float*)d_in[6];
  float* out = (float*)d_out;

  char* ws = (char*)d_ws;
  size_t off = 0;
  auto carve = [&](size_t bytes) -> void* {
    void* p = ws + off;
    off += (bytes + 255) & ~(size_t)255;
    return p;
  };
  const size_t nBSD = (size_t)BATCH * SEQ * D_MODEL;
  __bf16* xbf  = (__bf16*)carve(nBSD * 2);
  __bf16* xnbf = (__bf16*)carve(nBSD * 2);
  __bf16* wqb  = (__bf16*)carve((size_t)D_MODEL * D_MODEL * 2);
  __bf16* wkb  = (__bf16*)carve((size_t)D_MODEL * D_MODEL * 2);
  __bf16* wvb  = (__bf16*)carve((size_t)D_MODEL * D_MODEL * 2);
  __bf16* q_b  = (__bf16*)carve(nBSD * 2);          // [B,H,S,dk]
  __bf16* k_b  = (__bf16*)carve(nBSD * 2);          // [B,H,S,dk]
  __bf16* vt_b = (__bf16*)carve(nBSD * 2);          // [B,H,dk,S]
  unsigned char* msk = (unsigned char*)carve((size_t)BATCH * SEQ * SEQ);

  k_norm<<<BATCH * SEQ, 256, 0, stream>>>(x, xbf, xnbf);

  int nw = D_MODEL * D_MODEL;
  k_f2bf<<<nw / (256 * 4), 256, 0, stream>>>(Wq, wqb, nw);
  k_f2bf<<<nw / (256 * 4), 256, 0, stream>>>(Wk, wkb, nw);
  k_f2bf<<<nw / (256 * 4), 256, 0, stream>>>(Wv, wvb, nw);

  size_t shA = (size_t)64 * D_MODEL * sizeof(__bf16);       // 128KB TDM-staged tile
  dim3 gp(SEQ / 64, D_MODEL / 256, BATCH);
  k_proj<<<gp, 128, shA, stream>>>(xbf, wqb, bq, q_b, 0);
  k_proj<<<gp, 128, shA, stream>>>(xbf, wkb, bk, k_b, 0);
  k_proj<<<gp, 128, shA, stream>>>(xbf, wvb, bv, vt_b, 1);

  dim3 gm(SEQ / 64, SEQ / 256, BATCH);
  k_mask<<<gm, 128, shA, stream>>>(xnbf, msk);

  dim3 ga(SEQ / 16, NUM_HEADS, BATCH);
  size_t shAttn = (size_t)(16 * SEQ + 16) * sizeof(float)   // scores + rinv
                + (size_t)16 * D_K * sizeof(__bf16);        // Q tile
  k_attn<<<ga, 256, shAttn, stream>>>(q_b, k_b, vt_b, msk, out);
}
